// MotorDecoder_13572096655399
// MI455X (gfx1250) — compile-verified, matrix operand order checked by reading, not run on hardware
//
#include <hip/hip_runtime.h>

// ---------------------------------------------------------------------------
// MotorDecoder: 2-layer LSTM (H=512) x T=256 steps, B=256, out MLP 512->256->512
// Persistent WMMA kernel: 16 blocks x 16 batch rows, f16 WMMA / f32 accum.
// Independent WMMA chains (no hazard NOPs), bounded unrolling (no spills),
// single per-lane weight pointer + immediate displacements per fragment.
// ---------------------------------------------------------------------------

#define BATCH 256
#define PDIM  128
#define HDIM  512
#define TLEN  256
#define GDIM  2048        // 4*H
#define H2    256         // H/2
#define AROW  24          // padded halves per (kk,lane) row: 48B, bank-conflict free
#define TILEH (16 * 512)  // halves per N-tile in packed B layout (16 kk * 32 lanes * 16)

typedef __attribute__((ext_vector_type(16))) _Float16 v16h;
typedef __attribute__((ext_vector_type(8)))  float    v8f;

__device__ __forceinline__ float sigf(float x) { return 1.0f / (1.0f + __expf(-x)); }

// Map (row m, column col) of a 16 x K f16 A-matrix into the padded WMMA
// A-fragment half-index: layout = [kk (K/32)] [lane (32)] [AROW halves, 16 used]
// per ISA 7.12.2: v_r lanes0-15 hold K = 2(r%4)+h (+16 if r>=4); lanes16-31: +8
__device__ __forceinline__ int a_scatter_idx(int m, int col) {
    int kk    = col >> 5;
    int k32   = col & 31;
    int k16   = k32 & 15;
    int hi    = (k16 >> 3) & 1;           // which lane half
    int k8    = k16 & 7;
    int r     = (k8 >> 1) + ((k32 >> 4) << 2);
    int hh    = k8 & 1;
    int j     = (r << 1) + hh;
    int lane  = (m & 15) + (hi << 4);
    return (kk * 32 + lane) * AROW + j;
}

__device__ __forceinline__ v16h afrag(const _Float16* arr, int kk, int lane) {
    return *(const v16h*)(arr + (kk * 32 + lane) * AROW);
}

// ---------------------------------------------------------------------------
// Pack an N x K row-major fp32 matrix into f16 WMMA B-fragment tiles:
// layout [tileN][kk][lane][16 halves].
// ---------------------------------------------------------------------------
__global__ void pack_b_kernel(const float* __restrict__ src, _Float16* __restrict__ dst,
                              int N, int K) {
    int idx = blockIdx.x * blockDim.x + threadIdx.x;
    int total = N * K;
    if (idx >= total) return;
    int j    = idx & 15;
    int lane = (idx >> 4) & 31;
    int rest = idx >> 9;
    int kT   = K >> 5;
    int kk   = rest % kT;
    int nt   = rest / kT;
    int n    = nt * 16 + (lane & 15);
    int hi   = lane >> 4;
    int r    = j >> 1;
    int hh   = j & 1;
    int k32  = ((r & 3) << 1) + hh + (hi << 3) + ((r >> 2) << 4);
    int k    = kk * 32 + k32;
    dst[idx] = (_Float16)src[(size_t)n * K + k];
}

// h0 = (z+skill) @ Wh.T + bh ; c0 = (z+skill) @ Wc.T + bc   (shared by both layers)
__global__ void init_state_kernel(const float* __restrict__ z, const float* __restrict__ sk,
                                  const float* __restrict__ Wh, const float* __restrict__ bh,
                                  const float* __restrict__ Wc, const float* __restrict__ bc,
                                  float* __restrict__ h0, float* __restrict__ c0) {
    int idx = blockIdx.x * blockDim.x + threadIdx.x;
    if (idx >= BATCH * HDIM) return;
    int b = idx / HDIM, n = idx % HDIM;
    float sh = bh[n], sc = bc[n];
    const float* zr = z + (size_t)b * PDIM;
    const float* sr = sk + (size_t)b * PDIM;
    const float* wh = Wh + (size_t)n * PDIM;
    const float* wc = Wc + (size_t)n * PDIM;
    for (int k = 0; k < PDIM; ++k) {
        float cb = zr[k] + sr[k];
        sh = fmaf(cb, wh[k], sh);
        sc = fmaf(cb, wc[k], sc);
    }
    h0[idx] = sh; c0[idx] = sc;
}

__global__ void bias_sum_kernel(const float* a0, const float* b0,
                                const float* a1, const float* b1,
                                float* g0, float* g1) {
    int i = blockIdx.x * blockDim.x + threadIdx.x;
    if (i >= GDIM) return;
    g0[i] = a0[i] + b0[i];
    g1[i] = a1[i] + b1[i];
}

// ---------------------------------------------------------------------------
// One LSTM layer for this block's 16 batch rows. Wave w owns hidden-column
// tiles {2w, 2w+1}. Per kk: 2 passes (x-weights, h-weights) of 8 independent
// WMMAs, issued as 4-load/4-WMMA groups. Same-accumulator WMMA distance = 8
// (> the 5-slot CDNA5 WMMA->WMMA hazard window). All weight addresses derive
// from one per-lane pointer advanced 1KB/kk + compile-time displacements.
// ---------------------------------------------------------------------------
__device__ __forceinline__ void lstm_layer(const _Float16* __restrict__ xArr,
                                           const _Float16* __restrict__ hArr,
                                           _Float16* __restrict__ hOut,
                                           const _Float16* __restrict__ Wih,
                                           const _Float16* __restrict__ Whh,
                                           const float* __restrict__ bg,
                                           float creg[2][8], int wave, int lane) {
    int n16 = lane & 15, laneHi = lane >> 4;
    v8f acc[8];                                 // [ci*4 + g]
#pragma unroll
    for (int i = 0; i < 8; ++i) {
        int ci = i >> 2, g = i & 3;
        float b = bg[g * HDIM + (wave * 2 + ci) * 16 + n16];
#pragma unroll
        for (int v = 0; v < 8; ++v) acc[i][v] = b;
    }
    // per-lane base pointers at tile (wave*2), kk=0
    const _Float16* wx = Wih + (size_t)(wave * 2) * TILEH + lane * 16;
    const _Float16* wh = Whh + (size_t)(wave * 2) * TILEH + lane * 16;
#pragma unroll 1
    for (int kk = 0; kk < 16; ++kk) {
        v16h ax = afrag(xArr, kk, lane);
#pragma unroll
        for (int half = 0; half < 2; ++half) {      // pass 1: x @ W_ih
            v16h b0 = *(const v16h*)(wx + (size_t)((half * 2 + 0) * 32) * TILEH);
            v16h b1 = *(const v16h*)(wx + (size_t)((half * 2 + 0) * 32 + 1) * TILEH);
            v16h b2 = *(const v16h*)(wx + (size_t)((half * 2 + 1) * 32) * TILEH);
            v16h b3 = *(const v16h*)(wx + (size_t)((half * 2 + 1) * 32 + 1) * TILEH);
            acc[0 + (half * 2 + 0)] = __builtin_amdgcn_wmma_f32_16x16x32_f16(
                false, ax, false, b0, (short)0, acc[0 + (half * 2 + 0)], false, false);
            acc[4 + (half * 2 + 0)] = __builtin_amdgcn_wmma_f32_16x16x32_f16(
                false, ax, false, b1, (short)0, acc[4 + (half * 2 + 0)], false, false);
            acc[0 + (half * 2 + 1)] = __builtin_amdgcn_wmma_f32_16x16x32_f16(
                false, ax, false, b2, (short)0, acc[0 + (half * 2 + 1)], false, false);
            acc[4 + (half * 2 + 1)] = __builtin_amdgcn_wmma_f32_16x16x32_f16(
                false, ax, false, b3, (short)0, acc[4 + (half * 2 + 1)], false, false);
        }
        v16h ah = afrag(hArr, kk, lane);
#pragma unroll
        for (int half = 0; half < 2; ++half) {      // pass 2: h @ W_hh
            v16h b0 = *(const v16h*)(wh + (size_t)((half * 2 + 0) * 32) * TILEH);
            v16h b1 = *(const v16h*)(wh + (size_t)((half * 2 + 0) * 32 + 1) * TILEH);
            v16h b2 = *(const v16h*)(wh + (size_t)((half * 2 + 1) * 32) * TILEH);
            v16h b3 = *(const v16h*)(wh + (size_t)((half * 2 + 1) * 32 + 1) * TILEH);
            acc[0 + (half * 2 + 0)] = __builtin_amdgcn_wmma_f32_16x16x32_f16(
                false, ah, false, b0, (short)0, acc[0 + (half * 2 + 0)], false, false);
            acc[4 + (half * 2 + 0)] = __builtin_amdgcn_wmma_f32_16x16x32_f16(
                false, ah, false, b1, (short)0, acc[4 + (half * 2 + 0)], false, false);
            acc[0 + (half * 2 + 1)] = __builtin_amdgcn_wmma_f32_16x16x32_f16(
                false, ah, false, b2, (short)0, acc[0 + (half * 2 + 1)], false, false);
            acc[4 + (half * 2 + 1)] = __builtin_amdgcn_wmma_f32_16x16x32_f16(
                false, ah, false, b3, (short)0, acc[4 + (half * 2 + 1)], false, false);
        }
        wx += 512;      // advance one kk (1KB)
        wh += 512;
    }
    // acc index mapping above: acc[ci*4+g] with i = g for ci=0 (0..3), 4+g for ci=1.
#pragma unroll
    for (int ci = 0; ci < 2; ++ci) {
        int ct = wave * 2 + ci;
#pragma unroll
        for (int v = 0; v < 8; ++v) {
            float ig = sigf(acc[ci * 4 + 0][v]);
            float fg = sigf(acc[ci * 4 + 1][v]);
            float gg = tanhf(acc[ci * 4 + 2][v]);
            float og = sigf(acc[ci * 4 + 3][v]);
            float cn = fg * creg[ci][v] + ig * gg;
            creg[ci][v] = cn;
            float hn = og * tanhf(cn);
            int m = v + (laneHi << 3);
            hOut[a_scatter_idx(m, ct * 16 + n16)] = (_Float16)hn;
        }
    }
}

__global__ __launch_bounds__(512, 1) void motor_decoder_main(
    const _Float16* __restrict__ Wih0, const _Float16* __restrict__ Whh0,
    const _Float16* __restrict__ Wih1, const _Float16* __restrict__ Whh1,
    const _Float16* __restrict__ W1p,  const _Float16* __restrict__ W2p,
    const float* __restrict__ bg0, const float* __restrict__ bg1,
    const float* __restrict__ b1v, const float* __restrict__ b2v,
    const float* __restrict__ h0g, const float* __restrict__ c0g,
    float* __restrict__ outg) {
    // LDS state in padded WMMA A-fragment layout (16 kk x 32 lanes x AROW halves)
    __shared__ __align__(64) _Float16 xA[16 * 32 * AROW];       // 24KB
    __shared__ __align__(64) _Float16 h1A[2][16 * 32 * AROW];   // 48KB
    __shared__ __align__(64) _Float16 h2A[2][16 * 32 * AROW];   // 48KB
    __shared__ __align__(64) _Float16 midA[8 * 32 * AROW];      // 12KB

    int tid = threadIdx.x;
    int wave = tid >> 5, lane = tid & 31;
    int n16 = lane & 15, laneHi = lane >> 4;
    int m0 = blockIdx.x * 16;

    // ---- init: h0 -> both h buffers (fragment layout), x = 0 -----------------
    for (int e = tid; e < 16 * 32 * AROW; e += 512) xA[e] = (_Float16)0.0f;
    for (int e = tid; e < 8192; e += 512) {
        int m = e >> 9, col = e & 511;
        _Float16 hv = (_Float16)h0g[(size_t)(m0 + m) * HDIM + col];
        int si = a_scatter_idx(m, col);
        h1A[0][si] = hv;
        h2A[0][si] = hv;
    }
    // ---- c state: lives in VGPRs for the whole sequence ----------------------
    float c1r[2][8], c2r[2][8];
#pragma unroll
    for (int ci = 0; ci < 2; ++ci) {
        int ct = wave * 2 + ci;
#pragma unroll
        for (int v = 0; v < 8; ++v) {
            int m = v + (laneHi << 3);
            float cv = c0g[(size_t)(m0 + m) * HDIM + ct * 16 + n16];
            c1r[ci][v] = cv;
            c2r[ci][v] = cv;
        }
    }
    __syncthreads();

    for (int t = 0; t < TLEN; ++t) {
        int rb = t & 1, wb = rb ^ 1;

        // layer 0: gates = x @ Wih0.T + h1 @ Whh0.T + bg0
        lstm_layer(xA, h1A[rb], h1A[wb], Wih0, Whh0, bg0, c1r, wave, lane);
        __syncthreads();
        // layer 1: gates = h1 @ Wih1.T + h2 @ Whh1.T + bg1
        lstm_layer(h1A[wb], h2A[rb], h2A[wb], Wih1, Whh1, bg1, c2r, wave, lane);
        __syncthreads();

        // MLP fc1: mid = relu(h2 @ W1.T + b1) -> midA. 4 partial chains
        // (distance-4), bounded unroll so fragments are not spilled.
        {
            v8f pa[4];
            float b = b1v[wave * 16 + n16];
#pragma unroll
            for (int v = 0; v < 8; ++v) pa[0][v] = b;
#pragma unroll
            for (int p = 1; p < 4; ++p)
#pragma unroll
                for (int v = 0; v < 8; ++v) pa[p][v] = 0.0f;
            const _Float16* w1 = W1p + (size_t)wave * TILEH + lane * 16;
#pragma unroll 2
            for (int kq = 0; kq < 4; ++kq) {
#pragma unroll
                for (int p = 0; p < 4; ++p) {
                    int kk = kq * 4 + p;
                    v16h a = afrag(h2A[wb], kk, lane);
                    v16h bw = *(const v16h*)(w1 + (size_t)kk * 512);
                    pa[p] = __builtin_amdgcn_wmma_f32_16x16x32_f16(
                        false, a, false, bw, (short)0, pa[p], false, false);
                }
            }
#pragma unroll
            for (int v = 0; v < 8; ++v) {
                float s = (pa[0][v] + pa[1][v]) + (pa[2][v] + pa[3][v]);
                float r = fmaxf(s, 0.0f);
                int m = v + (laneHi << 3);
                midA[a_scatter_idx(m, wave * 16 + n16)] = (_Float16)r;
            }
        }
        __syncthreads();

        // MLP fc2: out = mid @ W2.T + b2 -> global fp32 + xA (next step input).
        // 2 column tiles x 2 K-halves = 4 independent chains.
        {
            v8f pa[2][2];
#pragma unroll
            for (int ci = 0; ci < 2; ++ci) {
                float b = b2v[(wave * 2 + ci) * 16 + n16];
#pragma unroll
                for (int v = 0; v < 8; ++v) { pa[ci][0][v] = b; pa[ci][1][v] = 0.0f; }
            }
            const _Float16* w2 = W2p + (size_t)(wave * 2) * (8 * 512) + lane * 16;
#pragma unroll 2
            for (int kq = 0; kq < 4; ++kq) {
                v16h alo = afrag(midA, kq, lane);
                v16h ahi = afrag(midA, kq + 4, lane);
                v16h b00 = *(const v16h*)(w2 + (size_t)kq * 512);
                v16h b10 = *(const v16h*)(w2 + (size_t)(8 + kq) * 512);
                pa[0][0] = __builtin_amdgcn_wmma_f32_16x16x32_f16(
                    false, alo, false, b00, (short)0, pa[0][0], false, false);
                pa[1][0] = __builtin_amdgcn_wmma_f32_16x16x32_f16(
                    false, alo, false, b10, (short)0, pa[1][0], false, false);
                v16h b01 = *(const v16h*)(w2 + (size_t)(4 + kq) * 512);
                v16h b11 = *(const v16h*)(w2 + (size_t)(12 + kq) * 512);
                pa[0][1] = __builtin_amdgcn_wmma_f32_16x16x32_f16(
                    false, ahi, false, b01, (short)0, pa[0][1], false, false);
                pa[1][1] = __builtin_amdgcn_wmma_f32_16x16x32_f16(
                    false, ahi, false, b11, (short)0, pa[1][1], false, false);
            }
#pragma unroll
            for (int ci = 0; ci < 2; ++ci) {
                int nt = wave * 2 + ci;
#pragma unroll
                for (int v = 0; v < 8; ++v) {
                    float s = pa[ci][0][v] + pa[ci][1][v];
                    int m = v + (laneHi << 3);
                    int col = nt * 16 + n16;
                    outg[((size_t)(m0 + m) * TLEN + t) * HDIM + col] = s;
                    xA[a_scatter_idx(m, col)] = (_Float16)s;
                }
            }
        }
        __syncthreads();
    }
}

// ---------------------------------------------------------------------------
extern "C" void kernel_launch(void* const* d_in, const int* in_sizes, int n_in,
                              void* d_out, int out_size, void* d_ws, size_t ws_size,
                              hipStream_t stream) {
    const float* z    = (const float*)d_in[0];
    const float* sk   = (const float*)d_in[1];
    const float* Wh   = (const float*)d_in[2];
    const float* bh   = (const float*)d_in[3];
    const float* Wc   = (const float*)d_in[4];
    const float* bc   = (const float*)d_in[5];
    const float* Wih0 = (const float*)d_in[6];
    const float* Whh0 = (const float*)d_in[7];
    const float* bih0 = (const float*)d_in[8];
    const float* bhh0 = (const float*)d_in[9];
    const float* Wih1 = (const float*)d_in[10];
    const float* Whh1 = (const float*)d_in[11];
    const float* bih1 = (const float*)d_in[12];
    const float* bhh1 = (const float*)d_in[13];
    const float* W1   = (const float*)d_in[14];
    const float* b1   = (const float*)d_in[15];
    const float* W2   = (const float*)d_in[16];
    const float* b2   = (const float*)d_in[17];
    float* outg = (float*)d_out;

    char* ws = (char*)d_ws;
    size_t o = 0;
    _Float16* Wih0p = (_Float16*)(ws + o); o += (size_t)GDIM * HDIM * 2;
    _Float16* Whh0p = (_Float16*)(ws + o); o += (size_t)GDIM * HDIM * 2;
    _Float16* Wih1p = (_Float16*)(ws + o); o += (size_t)GDIM * HDIM * 2;
    _Float16* Whh1p = (_Float16*)(ws + o); o += (size_t)GDIM * HDIM * 2;
    _Float16* W1p   = (_Float16*)(ws + o); o += (size_t)H2 * HDIM * 2;
    _Float16* W2p   = (_Float16*)(ws + o); o += (size_t)HDIM * H2 * 2;
    float* bg0 = (float*)(ws + o); o += GDIM * 4;
    float* bg1 = (float*)(ws + o); o += GDIM * 4;
    float* h0  = (float*)(ws + o); o += (size_t)BATCH * HDIM * 4;
    float* c0  = (float*)(ws + o); o += (size_t)BATCH * HDIM * 4;

    const int PB = 256;
    int nBig = GDIM * HDIM;   // 1,048,576 elements per matrix
    int gBig = (nBig + PB - 1) / PB;
    pack_b_kernel<<<gBig, PB, 0, stream>>>(Wih0, Wih0p, GDIM, HDIM);
    pack_b_kernel<<<gBig, PB, 0, stream>>>(Whh0, Whh0p, GDIM, HDIM);
    pack_b_kernel<<<gBig, PB, 0, stream>>>(Wih1, Wih1p, GDIM, HDIM);
    pack_b_kernel<<<gBig, PB, 0, stream>>>(Whh1, Whh1p, GDIM, HDIM);
    int nM1 = H2 * HDIM;
    pack_b_kernel<<<(nM1 + PB - 1) / PB, PB, 0, stream>>>(W1, W1p, H2, HDIM);
    int nM2 = HDIM * H2;
    pack_b_kernel<<<(nM2 + PB - 1) / PB, PB, 0, stream>>>(W2, W2p, HDIM, H2);

    bias_sum_kernel<<<(GDIM + PB - 1) / PB, PB, 0, stream>>>(bih0, bhh0, bih1, bhh1, bg0, bg1);

    int nInit = BATCH * HDIM;
    init_state_kernel<<<(nInit + PB - 1) / PB, PB, 0, stream>>>(z, sk, Wh, bh, Wc, bc, h0, c0);

    motor_decoder_main<<<BATCH / 16, 512, 0, stream>>>(
        Wih0p, Whh0p, Wih1p, Whh1p, W1p, W2p, bg0, bg1, b1, b2, h0, c0, outg);
}
